// MultiHeadAttention_83717502534076
// MI455X (gfx1250) — matrix operand order repaired and checked by reading, not compiled, not run on hardware
//
#include <hip/hip_runtime.h>
#include <hip/hip_bf16.h>

// ---------------------------------------------------------------------------
// MHA forward for MI455X (gfx1250, wave32, WMMA f16 -> f32 accumulate)
//   B=2, SQ=SKV=2048, D_MODEL=1024, H=16, Dh=64
// ---------------------------------------------------------------------------

typedef __attribute__((ext_vector_type(16))) _Float16 v16h;
typedef __attribute__((ext_vector_type(8)))  _Float16 v8h;
typedef __attribute__((ext_vector_type(8)))  float    v8f;

union HU {
    v16h v;
    v8h  h[2];
    _Float16 e[16];
};

static constexpr int B_  = 2;
static constexpr int H_  = 16;
static constexpr int SQ_ = 2048;
static constexpr int SKV_= 2048;
static constexpr int DM_ = 1024;
static constexpr int DH_ = 64;

__device__ __forceinline__ v8f wmma_f16(const v16h& a, const v16h& b, const v8f& c) {
    return __builtin_amdgcn_wmma_f32_16x16x32_f16(
        /*neg_a=*/false, a, /*neg_b=*/false, b,
        /*c_mod=*/(short)0, c, /*reuse_a=*/false, /*reuse_b=*/false);
}

// ---------------------------------------------------------------------------
// Generic tiled GEMM:  out(M=4096,N=1024) = A(4096,1024) * W(1024,1024) + bias
//   A_F32:    A is f32 (raw inputs) vs f16 (ctx from attention)
//   OUT_MODE: 0 = f32 row-major [M,N] (final output to d_out)
//             1 = f16 head-split [B,H,S,64] (Q,K to workspace)
//             2 = f16 head-split transposed [B,H,64,S] (V to workspace,
//                 so PV B-fragments are contiguous 32B global loads)
// Block: 256 threads (8 waves). Block tile 128(M) x 64(N). K chunk 32.
// ---------------------------------------------------------------------------
template <bool A_F32, int OUT_MODE>
__global__ __launch_bounds__(256)
void gemm_wmma(const void* __restrict__ Ap, const float* __restrict__ W,
               const float* __restrict__ bias, void* __restrict__ outp)
{
    constexpr int M = B_ * SQ_;   // 4096
    constexpr int N = DM_;        // 1024
    constexpr int K = DM_;        // 1024

    __shared__ _Float16 Alds[128 * 32];   // [row][k] row-major, stride 32
    __shared__ _Float16 Bt  [64 * 32];    // [col][k] (W transposed), stride 32

    const int t    = threadIdx.x;
    const int lane = t & 31;
    const int wave = t >> 5;
    const int half = lane >> 4;     // 0/1: which 16-lane half
    const int l16  = lane & 15;
    const int m0   = blockIdx.y * 128;
    const int n0   = blockIdx.x * 64;
    const int wm   = (wave >> 1) * 32;   // wave M offset in tile
    const int wn   = (wave & 1) * 32;    // wave N offset in tile

    v8f acc[2][2] = {};

    for (int kk = 0; kk < K; kk += 32) {
        // ---- stage A tile (128 x 32) as f16 ----
        if (A_F32) {
            const float* A = (const float*)Ap;
            #pragma unroll
            for (int j = 0; j < 4; ++j) {
                int idx = t + j * 256;                 // 0..1023 float4s
                int row = idx >> 3, c4 = (idx & 7) * 4;
                const float4 f = *(const float4*)(A + (size_t)(m0 + row) * K + kk + c4);
                _Float16* d = &Alds[row * 32 + c4];
                d[0] = (_Float16)f.x; d[1] = (_Float16)f.y;
                d[2] = (_Float16)f.z; d[3] = (_Float16)f.w;
            }
        } else {
            const _Float16* A = (const _Float16*)Ap;
            #pragma unroll
            for (int j = 0; j < 2; ++j) {
                int idx = t + j * 256;                 // 0..511 v8h chunks
                int row = idx >> 2, c8 = (idx & 3) * 8;
                *(v8h*)&Alds[row * 32 + c8] =
                    *(const v8h*)(A + (size_t)(m0 + row) * K + kk + c8);
            }
        }
        // ---- stage W tile (32 x 64) transposed to Bt[col][k] ----
        #pragma unroll
        for (int j = 0; j < 2; ++j) {
            int idx = t + j * 256;                     // 0..511 float4s
            int r = idx >> 4, c4 = (idx & 15) * 4;
            const float4 f = *(const float4*)(W + (size_t)(kk + r) * N + n0 + c4);
            Bt[(c4 + 0) * 32 + r] = (_Float16)f.x;
            Bt[(c4 + 1) * 32 + r] = (_Float16)f.y;
            Bt[(c4 + 2) * 32 + r] = (_Float16)f.z;
            Bt[(c4 + 3) * 32 + r] = (_Float16)f.w;
        }
        // prefetch next k-chunk into GL2 (gfx1250 global_prefetch_b8)
        if (kk + 32 < K) {
            if (A_F32)
                __builtin_prefetch((const float*)Ap + (size_t)(m0 + (t >> 1)) * K + kk + 32, 0, 1);
            __builtin_prefetch(W + (size_t)(kk + 32 + (t >> 3)) * N + n0, 0, 1);
        }
        __syncthreads();

        // ---- fragments + 4 WMMAs ----
        HU a0, a1, b0, b1;
        const _Float16* r0 = &Alds[(wm + l16) * 32];
        const _Float16* r1 = &Alds[(wm + 16 + l16) * 32];
        a0.v = *(const v16h*)(r0 + half * 16);
        a1.v = *(const v16h*)(r1 + half * 16);
        b0.v = *(const v16h*)&Bt[(wn + l16) * 32 + half * 16];
        b1.v = *(const v16h*)&Bt[(wn + 16 + l16) * 32 + half * 16];

        acc[0][0] = wmma_f16(a0.v, b0.v, acc[0][0]);
        acc[0][1] = wmma_f16(a0.v, b1.v, acc[0][1]);
        acc[1][0] = wmma_f16(a1.v, b0.v, acc[1][0]);
        acc[1][1] = wmma_f16(a1.v, b1.v, acc[1][1]);
        __syncthreads();
    }

    // ---- epilogue: bias + store ----
    #pragma unroll
    for (int i = 0; i < 2; ++i)
        #pragma unroll
        for (int j = 0; j < 2; ++j)
            #pragma unroll
            for (int g = 0; g < 8; ++g) {
                int R  = m0 + wm + i * 16 + g + half * 8;   // output row
                int Cc = n0 + wn + j * 16 + l16;            // output col
                float v = acc[i][j][g] + bias[Cc];
                if (OUT_MODE == 1) {
                    int b = R >> 11, s = R & (SQ_ - 1);
                    int h = Cc >> 6, d = Cc & 63;
                    ((_Float16*)outp)[(((size_t)b * H_ + h) * SQ_ + s) * DH_ + d] = (_Float16)v;
                } else if (OUT_MODE == 2) {
                    int b = R >> 11, s = R & (SQ_ - 1);
                    int h = Cc >> 6, d = Cc & 63;
                    ((_Float16*)outp)[(((size_t)b * H_ + h) * DH_ + d) * SKV_ + s] = (_Float16)v;
                } else {
                    ((float*)outp)[(size_t)R * N + Cc] = v;
                }
            }
}

// ---------------------------------------------------------------------------
// Attention core. Block = 128 threads (4 waves); block owns 64 q-rows of one
// (b,h); each wave owns 16 q-rows. K staged in 32-row LDS chunks shared by
// all waves; V read directly from global in transposed [B,H,64,SKV] layout.
// Pass 1: per-lane online row max/sum (no shuffles in the loop), single
// cross-lane merge afterwards. Pass 2: normalized weights to d_out, P@V via
// WMMA, context stored f16 as [B*SQ, 1024].
// ---------------------------------------------------------------------------
__global__ __launch_bounds__(128)
void attn_wmma(const _Float16* __restrict__ Qh, const _Float16* __restrict__ Kh,
               const _Float16* __restrict__ Vt, const int* __restrict__ mask,
               float* __restrict__ attnW, _Float16* __restrict__ ctx)
{
    __shared__ _Float16 Klds[32 * 64];
    __shared__ _Float16 Plds[4][16 * 32];   // per-wave P transpose scratch

    const int t    = threadIdx.x;
    const int lane = t & 31;
    const int wave = t >> 5;
    const int half = lane >> 4;
    const int l16  = lane & 15;
    const int bh   = blockIdx.y;            // b*16 + h
    const int b    = bh >> 4;
    const int h    = bh & 15;
    const int q0   = blockIdx.x * 64 + wave * 16;

    const _Float16* Kbase  = Kh + (size_t)bh * SKV_ * DH_;
    const _Float16* Vtbase = Vt + (size_t)bh * DH_ * SKV_;   // [64][SKV]

    // Q fragments (A layout, 16 rows x 64 dims = two K=32 fragments)
    HU qa0, qa1;
    {
        const _Float16* qrow = Qh + ((size_t)bh * SQ_ + q0 + l16) * DH_;
        qa0.v = *(const v16h*)(qrow + half * 16);
        qa1.v = *(const v16h*)(qrow + 32 + half * 16);
    }

    // per-lane online softmax state (lane owns cols == l16 mod 16)
    float lm[8], ll[8];
    #pragma unroll
    for (int g = 0; g < 8; ++g) { lm[g] = -INFINITY; ll[g] = 0.0f; }

    // ---------------- pass 1: per-lane online max + sum ----------------
    for (int kv0 = 0; kv0 < SKV_; kv0 += 32) {
        {   // stage K chunk (32 x 64)
            int row = t >> 2, c = (t & 3) * 16;
            *(v16h*)&Klds[row * 64 + c] =
                *(const v16h*)(Kbase + (size_t)(kv0 + row) * DH_ + c);
        }
        __syncthreads();
        #pragma unroll
        for (int sub = 0; sub < 2; ++sub) {
            const int c0 = sub * 16;
            HU kb0, kb1;   // B fragments: col = kv index, K = head dim
            kb0.v = *(const v16h*)&Klds[(c0 + l16) * 64 + half * 16];
            kb1.v = *(const v16h*)&Klds[(c0 + l16) * 64 + 32 + half * 16];
            v8f s = {};
            s = wmma_f16(qa0.v, kb0.v, s);
            s = wmma_f16(qa1.v, kb1.v, s);
            #pragma unroll
            for (int g = 0; g < 8; ++g) {
                int qr = q0 + g + half * 8;
                int kc = kv0 + c0 + l16;
                int mk = mask[((size_t)b * SQ_ + qr) * SKV_ + kc];
                float sv = (mk == 0) ? -1e9f : s[g] * 0.125f;
                float mo = lm[g];
                float mn = fmaxf(mo, sv);
                ll[g] = ll[g] * __expf(mo - mn) + __expf(sv - mn);
                lm[g] = mn;
            }
        }
        __syncthreads();
    }

    // single cross-lane merge of (max, sum) over the 16-lane row groups
    float m[8], rl[8];
    #pragma unroll
    for (int g = 0; g < 8; ++g) {
        float mo = lm[g], lo = ll[g];
        #pragma unroll
        for (int d = 1; d < 16; d <<= 1) {
            float mo2 = __shfl_xor(mo, d);
            float lo2 = __shfl_xor(lo, d);
            float mn  = fmaxf(mo, mo2);
            lo = lo * __expf(mo - mn) + lo2 * __expf(mo2 - mn);
            mo = mn;
        }
        m[g]  = mo;
        rl[g] = 1.0f / lo;
    }

    // ---------------- pass 2: weights out + P@V ----------------
    v8f o[4] = {};
    for (int kv0 = 0; kv0 < SKV_; kv0 += 32) {
        {   // stage K chunk
            int row = t >> 2, c = (t & 3) * 16;
            *(v16h*)&Klds[row * 64 + c] =
                *(const v16h*)(Kbase + (size_t)(kv0 + row) * DH_ + c);
        }
        __syncthreads();
        _Float16* Pw = Plds[wave];
        #pragma unroll
        for (int sub = 0; sub < 2; ++sub) {
            const int c0 = sub * 16;
            HU kb0, kb1;
            kb0.v = *(const v16h*)&Klds[(c0 + l16) * 64 + half * 16];
            kb1.v = *(const v16h*)&Klds[(c0 + l16) * 64 + 32 + half * 16];
            v8f s = {};
            s = wmma_f16(qa0.v, kb0.v, s);
            s = wmma_f16(qa1.v, kb1.v, s);
            #pragma unroll
            for (int g = 0; g < 8; ++g) {
                int qr = q0 + g + half * 8;
                int kc = kv0 + c0 + l16;
                int mk = mask[((size_t)b * SQ_ + qr) * SKV_ + kc];
                float sv = (mk == 0) ? -1e9f : s[g] * 0.125f;
                float p  = __expf(sv - m[g]) * rl[g];
                attnW[((size_t)bh * SQ_ + qr) * SKV_ + kc] = p;       // weights
                Pw[(g + half * 8) * 32 + c0 + l16] = (_Float16)p;     // transpose
            }
        }
        // wave-private LDS C-layout -> A-layout handoff; DS is in-order per
        // wave, make the dependency explicit:
        asm volatile("s_wait_dscnt 0" ::: "memory");

        HU pa;   // P as A fragment: 16 q-rows x K=32 kv
        const _Float16* prow = &Pw[l16 * 32];
        pa.v = *(const v16h*)(prow + half * 16);

        // V B-fragments: contiguous 32B global loads from transposed V
        #pragma unroll
        for (int nt = 0; nt < 4; ++nt) {
            HU vb;
            vb.v = *(const v16h*)(Vtbase + (size_t)(nt * 16 + l16) * SKV_
                                         + kv0 + half * 16);
            o[nt] = wmma_f16(pa.v, vb.v, o[nt]);
        }
        __syncthreads();
    }

    // ---- store context f16 as [B*SQ, 1024] (row-major for final GEMM) ----
    #pragma unroll
    for (int nt = 0; nt < 4; ++nt)
        #pragma unroll
        for (int g = 0; g < 8; ++g) {
            int qr = q0 + g + half * 8;
            int d  = h * DH_ + nt * 16 + l16;
            ctx[((size_t)b * SQ_ + qr) * DM_ + d] = (_Float16)(o[nt][g]);
        }
}

// ---------------------------------------------------------------------------
extern "C" void kernel_launch(void* const* d_in, const int* in_sizes, int n_in,
                              void* d_out, int out_size, void* d_ws, size_t ws_size,
                              hipStream_t stream) {
    const float* query = (const float*)d_in[0];
    const float* key   = (const float*)d_in[1];
    const float* value = (const float*)d_in[2];
    const int*   mask  = (const int*)d_in[3];
    const float* wq = (const float*)d_in[4];
    const float* bq = (const float*)d_in[5];
    const float* wk = (const float*)d_in[6];
    const float* bk = (const float*)d_in[7];
    const float* wv = (const float*)d_in[8];
    const float* bv = (const float*)d_in[9];
    const float* wo = (const float*)d_in[10];
    const float* bo = (const float*)d_in[11];

    const size_t headElems = (size_t)B_ * H_ * SQ_ * DH_;   // 4,194,304 halves
    _Float16* Qh  = (_Float16*)d_ws;
    _Float16* Kh  = Qh + headElems;
    _Float16* Vt  = Kh + headElems;                          // [B,H,64,SKV] f16
    _Float16* ctx = Vt + headElems;                          // [B*SQ, 1024] f16

    float* out   = (float*)d_out;                            // [B, SQ, 1024]
    float* attnW = out + (size_t)B_ * SQ_ * DM_;             // [B, H, SQ, SKV]

    dim3 gblk(256);
    dim3 ggrid(DM_ / 64, (B_ * SQ_) / 128);                  // 16 x 32 blocks

    gemm_wmma<true, 1><<<ggrid, gblk, 0, stream>>>(query, wq, bq, Qh);
    gemm_wmma<true, 1><<<ggrid, gblk, 0, stream>>>(key,   wk, bk, Kh);
    gemm_wmma<true, 2><<<ggrid, gblk, 0, stream>>>(value, wv, bv, Vt);

    dim3 ablk(128);
    dim3 agrid(SQ_ / 64, B_ * H_);                           // 32 x 32 blocks
    attn_wmma<<<agrid, ablk, 0, stream>>>(Qh, Kh, Vt, mask, attnW, ctx);

    gemm_wmma<false, 0><<<ggrid, gblk, 0, stream>>>(ctx, wo, bo, out);
}
